// SoftDTWDivergence_65403761984263
// MI455X (gfx1250) — matrix operand order repaired and checked by reading, not compiled, not run on hardware
//
#include <hip/hip_runtime.h>
#include <math.h>

typedef __attribute__((ext_vector_type(16))) _Float16 v16h;
typedef __attribute__((ext_vector_type(8)))  float    v8f;

#define BIG_F 1e30f
#define NB    32      // batches
#define TT    512     // sequence length
#define DD    64      // feature dim
#define ROWP  72      // padded f16 row stride in LDS (144 B, 16B aligned, spreads banks)

// d_ws layout (bytes):
//   xn f16 [32][512][64] @ 0        (2 MiB)
//   yn f16 [32][512][64] @ 2 MiB    (2 MiB)
//   partial f32 [96]     @ 4 MiB
#define WS_YN_OFF   (size_t)(NB * TT * DD * 2)
#define WS_PART_OFF (size_t)(2 * NB * TT * DD * 2)

// LDS layout (dynamic):
#define LDS_A_BYTES (TT * ROWP * 2)                 // 73728
#define LDS_B_BYTES (TT * ROWP * 2)                 // 73728
#define LDS_D_FLOATS (2 * 32 * 256)                 // 16384 floats = 65536 B
#define RSTRIDE 520
#define LDS_R_FLOATS (3 * RSTRIDE)                  // 1560 floats = 6240 B
#define SMEM_BYTES (LDS_A_BYTES + LDS_B_BYTES + LDS_D_FLOATS * 4 + LDS_R_FLOATS * 4)

// ---------------- Kernel 1: row L2-normalize, f32 -> f16 ----------------
__global__ void sdtw_norm_kernel(const float* __restrict__ x, const float* __restrict__ y,
                                 _Float16* __restrict__ xn, _Float16* __restrict__ yn) {
  int gwave = (int)((blockIdx.x * blockDim.x + threadIdx.x) >> 5); // one wave32 per row
  int lane  = (int)(threadIdx.x & 31);
  const float* src;
  _Float16*    dst;
  if (gwave < NB * TT) {
    src = x + (size_t)gwave * DD;
    dst = xn + (size_t)gwave * DD;
  } else {
    int r = gwave - NB * TT;
    src = y + (size_t)r * DD;
    dst = yn + (size_t)r * DD;
  }
  float2 v = ((const float2*)src)[lane];
  float s = v.x * v.x + v.y * v.y;
#pragma unroll
  for (int m = 16; m > 0; m >>= 1) s += __shfl_xor(s, m, 32);
  float scale = 1.0f / fmaxf(sqrtf(s), 1e-12f);
  dst[2 * lane]     = (_Float16)(v.x * scale);
  dst[2 * lane + 1] = (_Float16)(v.y * scale);
}

// ---------------- Kernel 2: fused WMMA distance tiles + soft-DTW DP ----------------
// grid: 96 blocks (m*32+b), block: 512 threads (16 waves). Thread t owns DP row t.
__global__ __launch_bounds__(512, 1)
void sdtw_dp_kernel(const _Float16* __restrict__ xn, const _Float16* __restrict__ yn,
                    float* __restrict__ partial) {
  extern __shared__ char smem[];
  _Float16* aS   = (_Float16*)smem;                       // [512][ROWP]
  _Float16* bS   = (_Float16*)(smem + LDS_A_BYTES);       // [512][ROWP]
  float*    Dbuf = (float*)(smem + LDS_A_BYTES + LDS_B_BYTES); // [2][32][16][16]
  float*    Rbuf = Dbuf + LDS_D_FLOATS;                   // [3][RSTRIDE]

  const int m = (int)blockIdx.x / NB;   // 0: xy, 1: xx, 2: yy
  const int b = (int)blockIdx.x % NB;
  const _Float16* Asrc = (m == 2) ? (yn + (size_t)b * TT * DD) : (xn + (size_t)b * TT * DD);
  const _Float16* Bsrc = (m == 1) ? (xn + (size_t)b * TT * DD) : (yn + (size_t)b * TT * DD);

  const int t = (int)threadIdx.x;

  // Stage normalized operands into padded LDS (row = 8 uint4 chunks).
  {
    const uint4* a4 = (const uint4*)Asrc;
    const uint4* b4 = (const uint4*)Bsrc;
    for (int k = t; k < TT * 8; k += 512) {
      int row = k >> 3, off = k & 7;
      ((uint4*)(aS + row * ROWP))[off] = a4[k];
      ((uint4*)(bS + row * ROWP))[off] = b4[k];
    }
  }
  __syncthreads();

  const int wave  = t >> 5;
  const int lane  = t & 31;
  const int laneM = lane & 15;   // M for A frag, N for B/C frags
  const int hi    = lane >> 4;

  float rprev = 0.0f;

  for (int g = 0; g < 64; ++g) {
    // ---- produce tile-diagonal g of D = 1 - A·B^T via WMMA into Dbuf[g&1] ----
    if (g <= 62) {
      int I0 = (g - 31 > 0) ? (g - 31) : 0;
      int I1 = (g < 31) ? g : 31;
      int cnt = I1 - I0 + 1;
#pragma unroll
      for (int rep = 0; rep < 2; ++rep) {
        int idx = wave + rep * 16;       // wave-uniform => EXEC all ones inside
        if (idx < cnt) {
          int I = I0 + idx, J = g - I;
          const _Float16* arow = aS + (16 * I + laneM) * ROWP;
          const _Float16* brow = bS + (16 * J + laneM) * ROWP;
          v8f acc = {};
#pragma unroll
          for (int kb = 0; kb < 2; ++kb) {
            int kbase = kb * 32;
            union { v16h v; uint4 q[2]; } A, Bm;
            // A 16x32 f16: per-lane chunks K=[8hi,8hi+8) and [16+8hi,16+8hi+8)
            A.q[0] = *(const uint4*)(arow + kbase + 8 * hi);
            A.q[1] = *(const uint4*)(arow + kbase + 16 + 8 * hi);
            // B 32x16 f16 (= yn rows): per-lane chunks K=[8hi,8hi+8) and [16+8hi,16+8hi+8)
            Bm.q[0] = *(const uint4*)(brow + kbase + 8 * hi);
            Bm.q[1] = *(const uint4*)(brow + kbase + 16 + 8 * hi);
            acc = __builtin_amdgcn_wmma_f32_16x16x32_f16(false, A.v, false, Bm.v,
                                                         (short)0, acc, false, false);
          }
          float* dtile = Dbuf + (g & 1) * 8192 + I * 256;
#pragma unroll
          for (int r8 = 0; r8 < 8; ++r8)
            dtile[(r8 + 8 * hi) * 16 + laneM] = 1.0f - acc[r8];
        }
      }
    }
    __syncthreads();

    // ---- 16 DP anti-diagonal steps: s = 16g + ss ----
    int sEnd = (g == 63) ? 15 : 16;  // last diagonal is s = 1022
    for (int ss = 0; ss < sEnd; ++ss) {
      int s = (g << 4) + ss;
      int c = s - t;                 // column of this thread's cell
      bool valid = (c >= 0) && (c < TT);
      float newv = 0.0f;
      if (valid) {
        // (s-1)%3 == (s+2)%3 ; (s-2)%3 == (s+1)%3
        float r_up   = (t == 0) ? BIG_F : Rbuf[((s + 2) % 3) * RSTRIDE + t];
        float r_diag = (t == 0) ? ((s == 0) ? 0.0f : BIG_F)
                                : ((c == 0) ? BIG_F : Rbuf[((s + 1) % 3) * RSTRIDE + t]);
        float r_left = (c == 0) ? BIG_F : rprev;
        int d = (t >> 4) + (c >> 4); // tile diagonal in {g-1, g}
        float Dv = Dbuf[(d & 1) * 8192 + (t >> 4) * 256 + (t & 15) * 16 + (c & 15)];
        float mn = fminf(r_diag, fminf(r_left, r_up));
        float sm = mn - __logf(__expf(mn - r_diag) + __expf(mn - r_left) + __expf(mn - r_up));
        newv = Dv + sm;
        Rbuf[(s % 3) * RSTRIDE + t + 1] = newv;  // buffer s%3 is untouched by this step's reads
        rprev = newv;
      }
      __syncthreads();
    }
  }

  if (t == TT - 1) partial[blockIdx.x] = rprev;  // R[512][512]
}

// ---------------- Kernel 3: combine partials ----------------
__global__ void sdtw_combine_kernel(const float* __restrict__ partial, float* __restrict__ out) {
  int b = (int)threadIdx.x;
  if (b < NB) out[b] = partial[b] - 0.5f * (partial[NB + b] + partial[2 * NB + b]);
}

extern "C" void kernel_launch(void* const* d_in, const int* in_sizes, int n_in,
                              void* d_out, int out_size, void* d_ws, size_t ws_size,
                              hipStream_t stream) {
  const float* x = (const float*)d_in[0];
  const float* y = (const float*)d_in[1];
  _Float16* xn = (_Float16*)d_ws;
  _Float16* yn = (_Float16*)((char*)d_ws + WS_YN_OFF);
  float* partial = (float*)((char*)d_ws + WS_PART_OFF);
  float* out = (float*)d_out;

  // 2 * 32 * 512 rows, one wave32 per row, 8 waves per 256-thread block
  sdtw_norm_kernel<<<(2 * NB * TT) / 8, 256, 0, stream>>>(x, y, xn, yn);

  (void)hipFuncSetAttribute((const void*)sdtw_dp_kernel,
                            hipFuncAttributeMaxDynamicSharedMemorySize, (int)SMEM_BYTES);
  sdtw_dp_kernel<<<3 * NB, 512, SMEM_BYTES, stream>>>(xn, yn, partial);

  sdtw_combine_kernel<<<1, 32, 0, stream>>>(partial, out);
}
